// NonLocalBlock_68083821576446
// MI455X (gfx1250) — matrix operand order repaired and checked by reading, compiled, not run on hardware
//
#include <hip/hip_runtime.h>

// ---------------------------------------------------------------------------
// NonLocalBlock for MI455X (gfx1250, wave32, WMMA + async global->LDS copies).
// B=32, C=1024, H=W=32, HW=1024, C_HALF=512.
//
// Pipeline (all bf16 WMMA GEMMs, f32 accumulate, f32 softmax):
//   0) xT  = transpose+cvt(x)            [b][n][c]   bf16
//   1) wq/wk/wv = cvt(Wq/Wk/Wv)                      bf16
//   2) qT  = (Wq*x)^T + bq               [b][n][o]   bf16   (GEMM, fused bias+transpose store)
//      kT  = (Wk*x)^T + bk               [b][m][o]   bf16
//      v   = (Wv*x)^T + bv               [b][n][o]   bf16
//   3) S   = qT * kT^T                   [b][n][m]   f32    (GEMM)
//   4) attn= softmax_rows(S)             [b][n][m]   bf16
//   5) out = v * attn^T + x              [b][i][j]   f32    (GEMM, fused residual)
// ---------------------------------------------------------------------------

typedef __bf16 bf16_t;
typedef __attribute__((ext_vector_type(16))) __bf16 v16bf;
typedef __attribute__((ext_vector_type(8)))  __bf16 v8bf;
typedef __attribute__((ext_vector_type(8)))  float  v8f;

union Frag16 { v16bf v; v8bf h[2]; };

#define BM 128
#define BN 128
#define BK 32
#define BKP 40   // padded LDS row (80B): conflict-free b128 reads

// --- CDNA5 async global->LDS copy (ASYNCcnt-tracked), with safe fallback ----
#if defined(__has_builtin)
#if __has_builtin(__builtin_amdgcn_s_wait_asynccnt) && defined(__gfx1250__)
#define USE_ASYNC_LDS 1
#endif
#endif

#if defined(USE_ASYNC_LDS)
// VGLOBAL async copy: LDS dest byte-address in a VGPR (low 32 bits of the
// generic pointer, per the aperture rule), 64-bit global address VGPR pair.
__device__ __forceinline__ void async_copy_b128(const void* g, void* l) {
    unsigned int       ldsa = (unsigned int)(uintptr_t)l;
    unsigned long long ga   = (unsigned long long)(uintptr_t)g;
    asm volatile("global_load_async_to_lds_b128 %0, %1, off"
                 :
                 : "v"(ldsa), "v"(ga)
                 : "memory");
}
#endif

// ---------------------------------------------------------------------------
// fp32 [b][c][n] -> bf16 [b][n][c] tiled transpose+convert
// ---------------------------------------------------------------------------
__global__ __launch_bounds__(256) void k_transpose_cvt(const float* __restrict__ x,
                                                       bf16_t* __restrict__ xT)
{
    __shared__ float tile[32][33];
    const int b  = blockIdx.z;
    const int n0 = blockIdx.x * 32;
    const int c0 = blockIdx.y * 32;
    const size_t base = (size_t)b * 1024 * 1024;
    const int tx = threadIdx.x, ty = threadIdx.y;
#pragma unroll
    for (int i = 0; i < 4; ++i)
        tile[ty + 8 * i][tx] = x[base + (size_t)(c0 + ty + 8 * i) * 1024 + (n0 + tx)];
    __syncthreads();
#pragma unroll
    for (int i = 0; i < 4; ++i)
        xT[base + (size_t)(n0 + ty + 8 * i) * 1024 + (c0 + tx)] =
            (bf16_t)tile[tx][ty + 8 * i];
}

// ---------------------------------------------------------------------------
// fp32 -> bf16 elementwise (weights); n is a multiple of 1024
// ---------------------------------------------------------------------------
__global__ __launch_bounds__(256) void k_cvt_bf16(const float* __restrict__ src,
                                                  bf16_t* __restrict__ dst, int n)
{
    int i = (blockIdx.x * 256 + threadIdx.x) * 4;
    if (i + 3 < n) {
        float4 v = *(const float4*)(src + i);
        dst[i + 0] = (bf16_t)v.x;
        dst[i + 1] = (bf16_t)v.y;
        dst[i + 2] = (bf16_t)v.z;
        dst[i + 3] = (bf16_t)v.w;
    }
}

// ---------------------------------------------------------------------------
// Generic batched bf16 WMMA GEMM:  C[M,N] = A[M,K] * BT[N,K]^T
//   - 256 threads = 8 wave32; block tile 128x128x32, wave tile 32x64
//   - double-buffered LDS, padded rows, async global->LDS staging
//   - optional fused bias (per output row), transposed-bf16 store, residual
// ---------------------------------------------------------------------------
template <bool TRANS_BF16_OUT, bool BIAS, bool RESID>
__global__ __launch_bounds__(256) void k_gemm_bf16(
    const bf16_t* __restrict__ A, const bf16_t* __restrict__ BT,
    const float* __restrict__ bias, const float* __restrict__ resid,
    float* __restrict__ outF, bf16_t* __restrict__ outB,
    int M, int N, int K,
    long aBatch, long bBatch, long oBatch, long rBatch)
{
    __shared__ __attribute__((aligned(16))) bf16_t lA[2][BM][BKP];
    __shared__ __attribute__((aligned(16))) bf16_t lB[2][BN][BKP];

    const int b  = blockIdx.z;
    const int m0 = blockIdx.y * BM;
    const int n0 = blockIdx.x * BN;
    const bf16_t* Ab = A + (size_t)b * aBatch;
    const bf16_t* Bb = BT + (size_t)b * bBatch;

    const int tid  = threadIdx.x;
    const int lane = tid & 31;
    const int wave = tid >> 5;
    const int wm   = (wave >> 1) * 32;   // 4 waves along M
    const int wn   = (wave & 1) * 64;    // 2 waves along N
    const int r16  = lane & 15;
    const int hi   = lane >> 4;          // half-wave select

    const v8f vzero = {0.f, 0.f, 0.f, 0.f, 0.f, 0.f, 0.f, 0.f};
    v8f acc[2][4];
#pragma unroll
    for (int i = 0; i < 2; ++i)
#pragma unroll
        for (int j = 0; j < 4; ++j) acc[i][j] = vzero;

    const int nk = K / BK;

    auto loadTile = [&](int kt, int bufi) {
#pragma unroll
        for (int i = 0; i < 2; ++i) {
            int c   = tid + i * 256;      // 512 chunks of 16B per matrix
            int row = c >> 2;
            int kc  = (c & 3) * 8;
            const bf16_t* gA = Ab + (size_t)(m0 + row) * K + (size_t)kt * BK + kc;
            const bf16_t* gB = Bb + (size_t)(n0 + row) * K + (size_t)kt * BK + kc;
#if defined(USE_ASYNC_LDS)
            async_copy_b128(gA, &lA[bufi][row][kc]);
            async_copy_b128(gB, &lB[bufi][row][kc]);
#else
            *(v8bf*)&lA[bufi][row][kc] = *(const v8bf*)gA;
            *(v8bf*)&lB[bufi][row][kc] = *(const v8bf*)gB;
#endif
        }
    };

    loadTile(0, 0);
#if defined(USE_ASYNC_LDS)
    __builtin_amdgcn_s_wait_asynccnt(0);
#endif
    __syncthreads();

    int buf = 0;
    for (int kt = 0; kt < nk; ++kt) {
        if (kt + 1 < nk) loadTile(kt + 1, buf ^ 1);
        if (kt + 2 < nk) {   // L2 prefetch two tiles ahead -> global_prefetch_b8
            __builtin_prefetch(Ab + (size_t)(m0 + (tid >> 1)) * K + (size_t)(kt + 2) * BK, 0, 1);
            __builtin_prefetch(Bb + (size_t)(n0 + (tid >> 1)) * K + (size_t)(kt + 2) * BK, 0, 1);
        }

        // A frags: lane holds row (l&15); K = hi*8+{0..7} and 16+hi*8+{0..7}
        Frag16 af[2];
#pragma unroll
        for (int i = 0; i < 2; ++i) {
            int row = wm + i * 16 + r16;
            af[i].h[0] = *(const v8bf*)&lA[buf][row][hi * 8];
            af[i].h[1] = *(const v8bf*)&lA[buf][row][16 + hi * 8];
        }
        // B frags: lane holds col (l&15); K = hi*8+{0..7} and 16+hi*8+{0..7}
        Frag16 bfr[4];
#pragma unroll
        for (int j = 0; j < 4; ++j) {
            int col = wn + j * 16 + r16;
            bfr[j].h[0] = *(const v8bf*)&lB[buf][col][hi * 8];
            bfr[j].h[1] = *(const v8bf*)&lB[buf][col][16 + hi * 8];
        }

#pragma unroll
        for (int i = 0; i < 2; ++i)
#pragma unroll
            for (int j = 0; j < 4; ++j)
                acc[i][j] = __builtin_amdgcn_wmma_f32_16x16x32_bf16(
                    false, af[i].v, false, bfr[j].v, (short)0, acc[i][j],
                    false, false);

#if defined(USE_ASYNC_LDS)
        // own async writes to lds[buf^1] must land before the barrier
        __builtin_amdgcn_s_wait_asynccnt(0);
#endif
        __syncthreads();
        buf ^= 1;
    }

    // Epilogue. C layout: VGPR r -> m = r + hi*8, n = lane&15 within 16x16 tile.
#pragma unroll
    for (int i = 0; i < 2; ++i)
#pragma unroll
        for (int j = 0; j < 4; ++j)
#pragma unroll
            for (int r = 0; r < 8; ++r) {
                int gm = m0 + wm + i * 16 + hi * 8 + r;
                int gn = n0 + wn + j * 16 + r16;
                float v = acc[i][j][r];
                if (BIAS) v += bias[gm];
                if (TRANS_BF16_OUT) {
                    outB[(size_t)b * oBatch + (size_t)gn * M + gm] = (bf16_t)v;
                } else {
                    if (RESID) v += resid[(size_t)b * rBatch + (size_t)gm * N + gn];
                    outF[(size_t)b * oBatch + (size_t)gm * N + gn] = v;
                }
            }
}

// ---------------------------------------------------------------------------
// Row softmax: one 256-thread block per row of 1024 f32, bf16 output
// ---------------------------------------------------------------------------
__global__ __launch_bounds__(256) void k_softmax(const float* __restrict__ S,
                                                 bf16_t* __restrict__ P, int N)
{
    __shared__ float red[256];
    const int row = blockIdx.x;
    const int b   = blockIdx.z;
    const float* s = S + ((size_t)b * 1024 + row) * N;
    bf16_t*      p = P + ((size_t)b * 1024 + row) * N;
    const int tid = threadIdx.x;

    float v[4];
    float m = -__builtin_inff();
#pragma unroll
    for (int i = 0; i < 4; ++i) { v[i] = s[tid + i * 256]; m = fmaxf(m, v[i]); }
    red[tid] = m;
    __syncthreads();
    for (int off = 128; off > 0; off >>= 1) {
        if (tid < off) red[tid] = fmaxf(red[tid], red[tid + off]);
        __syncthreads();
    }
    m = red[0];
    __syncthreads();

    float sum = 0.f;
#pragma unroll
    for (int i = 0; i < 4; ++i) { v[i] = __expf(v[i] - m); sum += v[i]; }
    red[tid] = sum;
    __syncthreads();
    for (int off = 128; off > 0; off >>= 1) {
        if (tid < off) red[tid] += red[tid + off];
        __syncthreads();
    }
    const float inv = 1.0f / red[0];
#pragma unroll
    for (int i = 0; i < 4; ++i) p[tid + i * 256] = (bf16_t)(v[i] * inv);
}

// ---------------------------------------------------------------------------
extern "C" void kernel_launch(void* const* d_in, const int* in_sizes, int n_in,
                              void* d_out, int out_size, void* d_ws, size_t ws_size,
                              hipStream_t stream)
{
    const float* x  = (const float*)d_in[0];
    const float* Wq = (const float*)d_in[1];
    const float* bq = (const float*)d_in[2];
    const float* Wk = (const float*)d_in[3];
    const float* bk = (const float*)d_in[4];
    const float* Wv = (const float*)d_in[5];
    const float* bv = (const float*)d_in[6];
    float* out = (float*)d_out;

    const int B = 32, C = 1024, HW = 1024, CH = 512;
    const size_t MB = 1ull << 20;
    char* ws = (char*)d_ws;

    bf16_t* xT = (bf16_t*)(ws + 0 * MB);    // 64 MB : [B][HW][C]
    bf16_t* wq = (bf16_t*)(ws + 64 * MB);   //  1 MB : [CH][C]
    bf16_t* wk = (bf16_t*)(ws + 65 * MB);   //  1 MB
    bf16_t* wv = (bf16_t*)(ws + 66 * MB);   //  2 MB : [C][C]
    bf16_t* qT = (bf16_t*)(ws + 68 * MB);   // 32 MB : [B][HW][CH]
    bf16_t* kT = (bf16_t*)(ws + 100 * MB);  // 32 MB : [B][HW][CH]
    bf16_t* vS = (bf16_t*)(ws + 132 * MB);  // 64 MB : [B][HW][C]
    float*  Sf = (float*)(ws + 196 * MB);   // 128 MB: [B][HW][HW]
    bf16_t* at = (bf16_t*)(ws + 324 * MB);  // 64 MB : [B][HW][HW]  (388 MB total)

    // 0) x -> xT (bf16, transposed)
    k_transpose_cvt<<<dim3(32, 32, B), dim3(32, 8), 0, stream>>>(x, xT);

    // 1) weight converts
    k_cvt_bf16<<<dim3(CH * C / 1024), dim3(256), 0, stream>>>(Wq, wq, CH * C);
    k_cvt_bf16<<<dim3(CH * C / 1024), dim3(256), 0, stream>>>(Wk, wk, CH * C);
    k_cvt_bf16<<<dim3(C * C / 1024), dim3(256), 0, stream>>>(Wv, wv, C * C);

    const long xB = (long)C * HW;      // xT batch stride
    const long qB = (long)HW * CH;     // qT/kT batch stride
    const long vB = (long)HW * C;      // v / S / attn / out batch stride

    // 2) q, k, v projections (bias fused, transposed bf16 store)
    k_gemm_bf16<true, true, false><<<dim3(HW / BN, CH / BM, B), dim3(256), 0, stream>>>(
        wq, xT, bq, nullptr, nullptr, qT, CH, HW, C, 0, xB, qB, 0);
    k_gemm_bf16<true, true, false><<<dim3(HW / BN, CH / BM, B), dim3(256), 0, stream>>>(
        wk, xT, bk, nullptr, nullptr, kT, CH, HW, C, 0, xB, qB, 0);
    k_gemm_bf16<true, true, false><<<dim3(HW / BN, C / BM, B), dim3(256), 0, stream>>>(
        wv, xT, bv, nullptr, nullptr, vS, C, HW, C, 0, xB, vB, 0);

    // 3) attention logits S = qT * kT^T
    k_gemm_bf16<false, false, false><<<dim3(HW / BN, HW / BM, B), dim3(256), 0, stream>>>(
        qT, kT, nullptr, nullptr, Sf, nullptr, HW, HW, CH, qB, qB, vB, 0);

    // 4) softmax rows -> attn (bf16)
    k_softmax<<<dim3(HW, 1, B), dim3(256), 0, stream>>>(Sf, at, HW);

    // 5) out = v * attn^T + x
    k_gemm_bf16<false, false, true><<<dim3(HW / BN, C / BM, B), dim3(256), 0, stream>>>(
        vS, at, nullptr, x, out, nullptr, C, HW, HW, vB, vB, vB, vB);

    (void)in_sizes; (void)n_in; (void)out_size; (void)ws_size;
}
